// UNetAttentionBlock_62302795596132
// MI455X (gfx1250) — compile-verified, matrix operand order checked by reading, not run on hardware
//
#include <hip/hip_runtime.h>
#include <hip/hip_bf16.h>
#include <math.h>

typedef __attribute__((ext_vector_type(16))) _Float16 v16h;
typedef __attribute__((ext_vector_type(8)))  _Float16 v8h;
typedef __attribute__((ext_vector_type(4)))  _Float16 v4h;
typedef __attribute__((ext_vector_type(8)))  float    v8f;
typedef __attribute__((ext_vector_type(4)))  int      v4i;

#define WMMA_F16(a, b, c) \
  __builtin_amdgcn_wmma_f32_16x16x32_f16(false, (a), false, (b), (short)0, (c), false, false)

// Async global->LDS copy (gfx1250 ASYNCcnt path), guarded so compile never breaks.
#if defined(__HIP_DEVICE_COMPILE__) && __has_builtin(__builtin_amdgcn_global_load_async_to_lds_b128)
#define USE_ASYNC_LDS 1
typedef __attribute__((address_space(1))) v4i* gptr_v4i;
typedef __attribute__((address_space(3))) v4i* lptr_v4i;
#else
#define USE_ASYNC_LDS 0
#endif

// ---------------------------------------------------------------------------
// Problem constants
// ---------------------------------------------------------------------------
#define BATCH 8
#define CHANS 512
#define SEQ   1024        // H*W = 32*32
#define NH    8
#define HD    64
#define OQKV  1536        // 3*C
#define PER_BATCH (CHANS * SEQ)   // 524288 = 2^19

// ---------------------------------------------------------------------------
// WMMA fragment loaders (CDNA5 wave32 layouts, CDNA5 ISA §7.12.2)
// A 16x32 f16: lane l -> row m = l&15, kh = l>>4; 16 K values:
//   j in [0,16): K = (j>=8 ? 16 : 0) + kh*8 + (j&7)
// B 32x16 f16: lane l -> col n = l&15, same K pattern along rows.
// C/D 16x16 f32: lane l, vgpr r -> m = r + 8*(l>>4), n = l&15.
// ---------------------------------------------------------------------------
__device__ __forceinline__ v16h load_a_frag(const _Float16* __restrict__ W,
                                            int ld, int row0, int k0) {
  int lane = threadIdx.x & 31;
  int m  = row0 + (lane & 15);
  int kh = (lane >> 4) & 1;
  const _Float16* p = W + (size_t)m * ld + k0 + kh * 8;
  union { unsigned int u[8]; v16h v; } r;
#pragma unroll
  for (int q = 0; q < 4; ++q)       // j = 0..7  -> K = kh*8 + j
    r.u[q] = *(const unsigned int*)(p + 2 * q);
#pragma unroll
  for (int q = 4; q < 8; ++q)       // j = 8..15 -> K = 16 + kh*8 + (j-8)
    r.u[q] = *(const unsigned int*)(p + 16 + 2 * (q - 4));
  return r.v;
}

// ---------------------------------------------------------------------------
// Kernel: per-batch mean / rstd over C*H*W (groupnorm with 1 group), float4 IO
// ---------------------------------------------------------------------------
__global__ __launch_bounds__(256) void stats_kernel(const float* __restrict__ src,
                                                    float* __restrict__ stats, int n) {
  __shared__ float s1[256];
  __shared__ float s2[256];
  int b = blockIdx.x;
  int tid = threadIdx.x;
  const float4* p = (const float4*)(src + (size_t)b * n);
  int n4 = n >> 2;
  float a = 0.f, q = 0.f;
  for (int i = tid; i < n4; i += 256) {
    float4 v = p[i];
    a += v.x + v.y + v.z + v.w;
    q += v.x * v.x + v.y * v.y + v.z * v.z + v.w * v.w;
  }
  s1[tid] = a;
  s2[tid] = q;
  __syncthreads();
  for (int s = 128; s > 0; s >>= 1) {
    if (tid < s) { s1[tid] += s1[tid + s]; s2[tid] += s2[tid + s]; }
    __syncthreads();
  }
  if (tid == 0) {
    float mean = s1[0] / (float)n;
    float var  = s2[0] / (float)n - mean * mean;
    stats[2 * b]     = mean;
    stats[2 * b + 1] = rsqrtf(var + 1e-5f);
  }
}

// ---------------------------------------------------------------------------
// Kernel: normalize x -> f16 (affine), 4 elements per thread
// ---------------------------------------------------------------------------
__global__ __launch_bounds__(256) void norm_to_h_kernel(const float* __restrict__ x,
                                                        const float* __restrict__ stats,
                                                        const float* __restrict__ w,
                                                        const float* __restrict__ bias,
                                                        _Float16* __restrict__ xh,
                                                        int total4) {
  int i = blockIdx.x * 256 + threadIdx.x;
  if (i >= total4) return;
  int base = i << 2;
  int b = base >> 19;              // / (512*1024)
  int c = (base >> 10) & 511;      // channel (4 | 1024 => same c for all 4)
  float k  = stats[2 * b + 1] * w[c];
  float bb = bias[c] - stats[2 * b] * k;
  float4 xv = ((const float4*)x)[i];
  v4h o;
  o[0] = (_Float16)(xv.x * k + bb);
  o[1] = (_Float16)(xv.y * k + bb);
  o[2] = (_Float16)(xv.z * k + bb);
  o[3] = (_Float16)(xv.w * k + bb);
  ((v4h*)xh)[i] = o;
}

// ---------------------------------------------------------------------------
// Kernel: f32 -> f16 weight conversion, 4 elements per thread
// ---------------------------------------------------------------------------
__global__ __launch_bounds__(256) void cvt_f32_to_f16_kernel(const float* __restrict__ src,
                                                             _Float16* __restrict__ dst,
                                                             int n4) {
  int i = blockIdx.x * 256 + threadIdx.x;
  if (i >= n4) return;
  float4 v = ((const float4*)src)[i];
  v4h o;
  o[0] = (_Float16)v.x; o[1] = (_Float16)v.y;
  o[2] = (_Float16)v.z; o[3] = (_Float16)v.w;
  ((v4h*)dst)[i] = o;
}

// ---------------------------------------------------------------------------
// Kernel: batched GEMM  out[b][o][s] = sum_c A[o][c] * B[b][c][s] + bias[o]
// block = 256 threads = 8 waves in a 2(M) x 4(N) grid; wave tile 32x32 (2x2 WMMA)
// block tile: 64 (O) x 128 (S). B tile (32 x 128) staged per K-step into LDS,
// pre-swizzled into B-fragment order -> per-wave operand reads are ds_load_b128.
// ---------------------------------------------------------------------------
template <bool OUT_F32>
__global__ __launch_bounds__(256) void gemm_wmma_kernel(const _Float16* __restrict__ A,
                                                        const _Float16* __restrict__ Bm,
                                                        const float* __restrict__ bias,
                                                        void* __restrict__ out,
                                                        int O, int C, int S) {
  __shared__ _Float16 bstage[4096];   // 8 KB: [nt(8)][lane(32)][j(16)]
  const int bidx = blockIdx.z;
  const _Float16* Bb = Bm + (size_t)bidx * C * S;
  const int sblk = blockIdx.x * 128;
  const int oblk = blockIdx.y * 64;
  const int tid  = threadIdx.x;
  const int wid  = tid >> 5;
  const int wr   = wid >> 2;       // 0..1
  const int wc   = wid & 3;        // 0..3
  const int ow   = oblk + wr * 32;
  const int lane = tid & 31;

  const v8f vzero = {0.f, 0.f, 0.f, 0.f, 0.f, 0.f, 0.f, 0.f};
  v8f acc[2][2];
#pragma unroll
  for (int mt = 0; mt < 2; ++mt)
#pragma unroll
    for (int nt = 0; nt < 2; ++nt) acc[mt][nt] = vzero;

  for (int k0 = 0; k0 < C; k0 += 32) {
    __syncthreads();   // previous iteration's reads done
    // ---- stage B tile 32(K) x 128(N) into fragment order ----
#pragma unroll
    for (int c2 = tid; c2 < 512; c2 += 256) {
      int K  = c2 >> 4;             // 0..31
      int n0 = (c2 & 15) << 3;      // 0..120
      v8h val = *(const v8h*)(Bb + (size_t)(k0 + K) * S + sblk + n0);
      int laneh = (K >> 3) & 1;
      int j = (K & 7) + 8 * ((K >> 4) & 1);
#pragma unroll
      for (int e = 0; e < 8; ++e) {
        int n = n0 + e;
        bstage[(((n >> 4) * 32) + laneh * 16 + (n & 15)) * 16 + j] = val[e];
      }
    }
    if (k0 + 32 < C) {  // hint next B tile into cache (global_prefetch_b8)
      __builtin_prefetch(Bb + (size_t)(k0 + 32 + (tid >> 4)) * S + sblk + (tid & 15) * 8, 0, 1);
    }
    __syncthreads();

    v16h a0 = load_a_frag(A, C, ow, k0);
    v16h a1 = load_a_frag(A, C, ow + 16, k0);
    v16h b0 = *(const v16h*)(bstage + ((wc * 2 + 0) * 32 + lane) * 16);
    v16h b1 = *(const v16h*)(bstage + ((wc * 2 + 1) * 32 + lane) * 16);
    acc[0][0] = WMMA_F16(a0, b0, acc[0][0]);
    acc[0][1] = WMMA_F16(a0, b1, acc[0][1]);
    acc[1][0] = WMMA_F16(a1, b0, acc[1][0]);
    acc[1][1] = WMMA_F16(a1, b1, acc[1][1]);
  }

  const int lh = (lane >> 4) & 1;
  const int ll = lane & 15;
  const int sw = sblk + wc * 32;
#pragma unroll
  for (int mt = 0; mt < 2; ++mt)
#pragma unroll
    for (int nt = 0; nt < 2; ++nt)
#pragma unroll
      for (int r = 0; r < 8; ++r) {
        int o = ow + mt * 16 + r + 8 * lh;
        int s = sw + nt * 16 + ll;
        float vv = acc[mt][nt][r] + bias[o];
        size_t idx = ((size_t)bidx * O + o) * S + s;
        if (OUT_F32) ((float*)out)[idx] = vv;
        else         ((_Float16*)out)[idx] = (_Float16)vv;
      }
}

// ---------------------------------------------------------------------------
// Kernel: attention. Grid (b*h, qslice): each block stages full per-head K,V
// (256 KB) into LDS in WMMA fragment order and processes 256 queries with
// 16 waves (one 16-query tile per wave), online-softmax on transposed score
// tiles S^T = K^T*Q so exp(S) lands directly in P's A-fragment layout.
//   kfrag[ktile(64)][dstep(2)][lane(32)][j(16)] : A-frags of K^T (rows=kpos, K=d)
//   vfrag[kc(32)][dt(4)][lane(32)][j(16)]       : B-frags of V   (K=kpos, cols=d)
// V's swizzle is a 16B-granular permutation -> staged with
// global_load_async_to_lds_b128 (ASYNCcnt) when available.
// ---------------------------------------------------------------------------
__global__ __launch_bounds__(512) void attn_kernel(const _Float16* __restrict__ qkv,
                                                   _Float16* __restrict__ outh) {
  extern __shared__ _Float16 smem[];
  _Float16* kfrag = smem;            // 65536 f16 = 128 KB
  _Float16* vfrag = smem + 65536;    // 65536 f16 = 128 KB

  const int bh = blockIdx.x;
  const int b = bh >> 3, h = bh & 7;
  const _Float16* base = qkv + (size_t)b * OQKV * SEQ;
  const _Float16* Qp = base + (size_t)(h * HD) * SEQ;
  const _Float16* Kp = base + (size_t)(CHANS + h * HD) * SEQ;
  const _Float16* Vp = base + (size_t)(2 * CHANS + h * HD) * SEQ;
  const int tid = threadIdx.x;

  // ---- Stage K and V (64 x 1024 f16 each) ----
  for (int i = tid; i < 8192; i += 512) {
    int lin = i << 3;                 // 8 f16 per thread-iteration
    int d = lin >> 10, s0 = lin & 1023;
    // K: genuine transpose (lane dim = kpos) -> VALU scatter, ds_store_b16 x8
    v8h kv = *(const v8h*)(Kp + (size_t)d * SEQ + s0);
    int d5 = d & 31, dstep = d >> 5;
    int lanehK = (d5 >> 3) & 1;
    int jK = (d5 & 7) + 8 * ((d5 >> 4) & 1);
#pragma unroll
    for (int e = 0; e < 8; ++e) {
      int s = s0 + e;
      int laneK = lanehK * 16 + (s & 15);
      kfrag[((((s >> 4) * 2 + dstep) * 32) + laneK) * 16 + jK] = kv[e];
    }
    // V: 16B-contiguous permutation -> async DMA straight into place
    {
      int kc = s0 >> 5, k5 = s0 & 31;          // k5 in {0,8,16,24}
      int lanehV = (k5 >> 3) & 1;
      int jbase  = (k5 >= 16) ? 8 : 0;
      int dtV = d >> 4;
      int laneV = lanehV * 16 + (d & 15);
      int dstIdx = (((kc * 4 + dtV) * 32) + laneV) * 16 + jbase;   // f16 units, 16B aligned
#if USE_ASYNC_LDS
      __builtin_amdgcn_global_load_async_to_lds_b128(
          (gptr_v4i)(Vp + (size_t)d * SEQ + s0),
          (lptr_v4i)(vfrag + dstIdx), 0, 0);
#else
      v8h vv = *(const v8h*)(Vp + (size_t)d * SEQ + s0);
      *(v8h*)(vfrag + dstIdx) = vv;
#endif
    }
  }
#if USE_ASYNC_LDS
#if __has_builtin(__builtin_amdgcn_s_wait_asynccnt)
  __builtin_amdgcn_s_wait_asynccnt(0);
#else
  asm volatile("s_wait_asynccnt 0" ::: "memory");
#endif
#endif
  __syncthreads();

  const int wid = tid >> 5;
  const int lane = tid & 31;
  const int lh = (lane >> 4) & 1;
  const int ll = lane & 15;
  const v8f vzero = {0.f, 0.f, 0.f, 0.f, 0.f, 0.f, 0.f, 0.f};
  const int qb = (blockIdx.y * 16 + wid) * 16;   // this wave's 16-query tile

  // Q fragments (B role: K=d, N=q), softmax scale folded in (hd^-0.5 = 0.125)
  v16h qf[2];
#pragma unroll
  for (int ds = 0; ds < 2; ++ds)
#pragma unroll
    for (int j = 0; j < 16; ++j) {
      int d = ds * 32 + ((j >= 8) ? 16 : 0) + lh * 8 + (j & 7);
      qf[ds][j] = (_Float16)((float)Qp[(size_t)d * SEQ + qb + ll] * 0.125f);
    }

  v8f acc[4];
#pragma unroll
  for (int dt = 0; dt < 4; ++dt) acc[dt] = vzero;
  float mrun = -3.0e38f;
  float lrun = 0.f;

  for (int kc = 0; kc < 32; ++kc) {
    // S^T tiles for kpos chunk [32*kc, 32*kc+32): two 16x16 D tiles
    v8f st0 = vzero, st1 = vzero;
#pragma unroll
    for (int ds = 0; ds < 2; ++ds) {
      v16h ka0 = *(const v16h*)(kfrag + ((((2 * kc)     * 2 + ds) * 32) + lane) * 16);
      v16h ka1 = *(const v16h*)(kfrag + ((((2 * kc + 1) * 2 + ds) * 32) + lane) * 16);
      st0 = WMMA_F16(ka0, qf[ds], st0);
      st1 = WMMA_F16(ka1, qf[ds], st1);
    }

    // online softmax for query q = ll (this lane's column of S^T)
    float cm = st0[0];
#pragma unroll
    for (int r = 0; r < 8; ++r) { cm = fmaxf(cm, st0[r]); cm = fmaxf(cm, st1[r]); }
    cm = fmaxf(cm, __shfl_xor(cm, 16, 32));
    float newm = fmaxf(mrun, cm);
    float corr = __expf(mrun - newm);
    float p0[8], p1[8];
    float csum = 0.f;
#pragma unroll
    for (int r = 0; r < 8; ++r) {
      p0[r] = __expf(st0[r] - newm);
      p1[r] = __expf(st1[r] - newm);
      csum += p0[r] + p1[r];
    }
    csum += __shfl_xor(csum, 16, 32);
    lrun = lrun * corr + csum;
    mrun = newm;

    // rescale accumulators: acc row q = r + 8*lh -> fetch that q's corr
    float corr_b[8];
#pragma unroll
    for (int r = 0; r < 8; ++r) corr_b[r] = __shfl(corr, r + 8 * lh, 32);
#pragma unroll
    for (int dt = 0; dt < 4; ++dt)
#pragma unroll
      for (int r = 0; r < 8; ++r) acc[dt][r] *= corr_b[r];

    // P^T D-layout == P A-fragment layout: pack in-lane, no shuffles
    v16h pf;
#pragma unroll
    for (int r = 0; r < 8; ++r) {
      pf[r]     = (_Float16)p0[r];
      pf[8 + r] = (_Float16)p1[r];
    }
#pragma unroll
    for (int dt = 0; dt < 4; ++dt) {
      v16h vfr = *(const v16h*)(vfrag + (((kc * 4 + dt) * 32) + lane) * 16);
      acc[dt] = WMMA_F16(pf, vfr, acc[dt]);
    }
  }

  // normalize by row sum and store: out channel c = h*64 + dt*16 + ll, s = q
  float invl = 1.0f / lrun;
  float inv_b[8];
#pragma unroll
  for (int r = 0; r < 8; ++r) inv_b[r] = __shfl(invl, r + 8 * lh, 32);
#pragma unroll
  for (int dt = 0; dt < 4; ++dt)
#pragma unroll
    for (int r = 0; r < 8; ++r) {
      int c = h * HD + dt * 16 + ll;
      int s = qb + r + 8 * lh;
      outh[(size_t)b * CHANS * SEQ + (size_t)c * SEQ + s] =
          (_Float16)(acc[dt][r] * inv_b[r]);
    }
}

// ---------------------------------------------------------------------------
// Kernel: final groupnorm + residual, 4 elements per thread
// ---------------------------------------------------------------------------
__global__ __launch_bounds__(256) void final_kernel(const float* __restrict__ p,
                                                    const float* __restrict__ stats,
                                                    const float* __restrict__ w,
                                                    const float* __restrict__ bias,
                                                    const float* __restrict__ x,
                                                    float* __restrict__ out,
                                                    int total4) {
  int i = blockIdx.x * 256 + threadIdx.x;
  if (i >= total4) return;
  int base = i << 2;
  int b = base >> 19;
  int c = (base >> 10) & 511;
  float k  = stats[2 * b + 1] * w[c];
  float bb = bias[c] - stats[2 * b] * k;
  float4 pv = ((const float4*)p)[i];
  float4 xv = ((const float4*)x)[i];
  float4 ov;
  ov.x = pv.x * k + bb + xv.x;
  ov.y = pv.y * k + bb + xv.y;
  ov.z = pv.z * k + bb + xv.z;
  ov.w = pv.w * k + bb + xv.w;
  ((float4*)out)[i] = ov;
}

// ---------------------------------------------------------------------------
// Host launcher
// ---------------------------------------------------------------------------
extern "C" void kernel_launch(void* const* d_in, const int* in_sizes, int n_in,
                              void* d_out, int out_size, void* d_ws, size_t ws_size,
                              hipStream_t stream) {
  (void)in_sizes; (void)n_in; (void)out_size; (void)ws_size;
  const float* x      = (const float*)d_in[0];
  const float* norm_w = (const float*)d_in[1];
  const float* norm_b = (const float*)d_in[2];
  const float* qkv_w  = (const float*)d_in[3];
  const float* qkv_b  = (const float*)d_in[4];
  const float* proj_w = (const float*)d_in[5];
  const float* proj_b = (const float*)d_in[6];
  const float* onw    = (const float*)d_in[7];
  const float* onb    = (const float*)d_in[8];
  float* out = (float*)d_out;

  // workspace carve-up
  char* ws = (char*)d_ws;
  size_t off = 0;
  float* stats1 = (float*)(ws + off); off += 256;
  float* stats2 = (float*)(ws + off); off += 256;
  _Float16* xh     = (_Float16*)(ws + off); off += (size_t)BATCH * PER_BATCH * 2;       // 8 MB
  _Float16* wqkvh  = (_Float16*)(ws + off); off += (size_t)OQKV * CHANS * 2;            // 1.5 MB
  _Float16* wprojh = (_Float16*)(ws + off); off += (size_t)CHANS * CHANS * 2;           // 0.5 MB
  _Float16* qkvh   = (_Float16*)(ws + off); off += (size_t)BATCH * OQKV * SEQ * 2;      // 24 MB
  _Float16* attnh  = (_Float16*)(ws + off); off += (size_t)BATCH * PER_BATCH * 2;       // 8 MB
  float*    projf  = (float*)(ws + off);    off += (size_t)BATCH * PER_BATCH * 4;       // 16 MB

  const int total  = BATCH * PER_BATCH;             // 4194304
  const int total4 = total / 4;

  // 1) groupnorm-1 stats + normalize to f16
  stats_kernel<<<BATCH, 256, 0, stream>>>(x, stats1, PER_BATCH);
  norm_to_h_kernel<<<(total4 + 255) / 256, 256, 0, stream>>>(x, stats1, norm_w, norm_b, xh, total4);

  // 2) weights to f16
  cvt_f32_to_f16_kernel<<<(OQKV * CHANS / 4 + 255) / 256, 256, 0, stream>>>(qkv_w, wqkvh, OQKV * CHANS / 4);
  cvt_f32_to_f16_kernel<<<(CHANS * CHANS / 4 + 255) / 256, 256, 0, stream>>>(proj_w, wprojh, CHANS * CHANS / 4);

  // 3) qkv GEMM: [1536 x 512] x [512 x 1024] per batch -> f16
  gemm_wmma_kernel<false><<<dim3(SEQ / 128, OQKV / 64, BATCH), 256, 0, stream>>>(
      wqkvh, xh, qkv_b, (void*)qkvh, OQKV, CHANS, SEQ);

  // 4) attention: grid (b*h, qslice); 512 threads; 256 KB dynamic LDS
  static_assert(2 * 65536 * sizeof(_Float16) == 262144, "lds size");
  (void)hipFuncSetAttribute((const void*)attn_kernel,
                            hipFuncAttributeMaxDynamicSharedMemorySize, 262144);
  attn_kernel<<<dim3(BATCH * NH, 4), 512, 262144, stream>>>(qkvh, attnh);

  // 5) proj GEMM: [512 x 512] x [512 x 1024] per batch -> f32
  gemm_wmma_kernel<true><<<dim3(SEQ / 128, CHANS / 64, BATCH), 256, 0, stream>>>(
      wprojh, attnh, proj_b, (void*)projf, CHANS, CHANS, SEQ);

  // 6) groupnorm-2 stats + final affine + residual
  stats_kernel<<<BATCH, 256, 0, stream>>>(projf, stats2, PER_BATCH);
  final_kernel<<<(total4 + 255) / 256, 256, 0, stream>>>(projf, stats2, onw, onb, x, out, total4);
}
